// DFRNN_17678085390828
// MI455X (gfx1250) — matrix operand order repaired and checked by reading, not compiled, not run on hardware
//
#include <hip/hip_runtime.h>
#include <hip/hip_bf16.h>

#define NP   256
#define NS   (NP * NP)
#define OSZ  768

typedef __attribute__((ext_vector_type(16))) _Float16 v16h;
typedef __attribute__((ext_vector_type(8)))  _Float16 v8h;
typedef __attribute__((ext_vector_type(8)))  float    v8f;

// ---------------- float <-> sortable key -------------------------------
__device__ __forceinline__ unsigned f2key(float v) {
    unsigned b = __float_as_uint(v);
    return (b & 0x80000000u) ? ~b : (b | 0x80000000u);
}
__device__ __forceinline__ float key2f(unsigned k) {
    unsigned b = (k & 0x80000000u) ? (k & 0x7FFFFFFFu) : ~k;
    return __uint_as_float(b);
}

// ---------------- block-cooperative radix select (k-th smallest) -------
__device__ float block_select(const float* __restrict__ data, int n, unsigned k,
                              unsigned* hist, unsigned* sh) {
    const int tid = threadIdx.x, nt = blockDim.x;
    if (tid == 0) { sh[0] = 0u; sh[1] = k; }
    __syncthreads();
    for (int shift = 24; shift >= 0; shift -= 8) {
        for (int b = tid; b < 256; b += nt) hist[b] = 0u;
        __syncthreads();
        unsigned prefix = sh[0];
        unsigned maskhi = (shift == 24) ? 0u : (0xFFFFFFFFu << (shift + 8));
        for (int i = tid; i < n; i += nt) {
            unsigned key = f2key(data[i]);
            if ((key & maskhi) == prefix)
                atomicAdd(&hist[(key >> shift) & 0xFFu], 1u);
        }
        __syncthreads();
        if (tid == 0) {
            unsigned kk = sh[1], run = 0u; int sel = 255;
            for (int b = 0; b < 256; ++b) {
                unsigned c = hist[b];
                if (kk < run + c) { sel = b; kk -= run; break; }
                run += c;
            }
            sh[0] = prefix | ((unsigned)sel << shift);
            sh[1] = kk;
        }
        __syncthreads();
    }
    return key2f(sh[0]);
}

// ---------------- bias stats: mean + median ----------------------------
__global__ __launch_bounds__(1024) void k_stats(const float* __restrict__ bias,
                                                float* __restrict__ wsf) {
    __shared__ float    rbuf[1024];
    __shared__ unsigned hist[256];
    __shared__ unsigned sh[2];
    const int tid = threadIdx.x;
    float s = 0.0f;
    for (int i = tid; i < NS; i += 1024) s += bias[i];
    rbuf[tid] = s;
    __syncthreads();
    for (int o = 512; o > 0; o >>= 1) {
        if (tid < o) rbuf[tid] += rbuf[tid + o];
        __syncthreads();
    }
    if (tid == 0) wsf[0] = rbuf[0] * (1.0f / NS);
    float m0 = block_select(bias, NS, NS / 2 - 1, hist, sh);
    float m1 = block_select(bias, NS, NS / 2,     hist, sh);
    if (tid == 0) wsf[1] = 0.5f * (m0 + m1);
}

// ---------------- mean of charge (per step) ----------------------------
__global__ __launch_bounds__(1024) void k_meanq(const float* __restrict__ charge,
                                                float* __restrict__ wsf) {
    __shared__ float rbuf[1024];
    const int tid = threadIdx.x;
    float s = 0.0f;
    for (int i = tid; i < NS; i += 1024) s += charge[i];
    rbuf[tid] = s;
    __syncthreads();
    for (int o = 512; o > 0; o >>= 1) {
        if (tid < o) rbuf[tid] += rbuf[tid + o];
        __syncthreads();
    }
    if (tid == 0) wsf[2] = rbuf[0] * (1.0f / NS);
}

// ---------------- downsample illum (sum) & sens (mean) -----------------
__global__ __launch_bounds__(256) void k_down(const float* __restrict__ ill,
                                              const float* __restrict__ sin_,
                                              float* __restrict__ illum,
                                              float* __restrict__ sens) {
    int p = blockIdx.x * blockDim.x + threadIdx.x;
    if (p >= NS) return;
    int i = p >> 8, j = p & 255;
    float si = 0.0f, ss = 0.0f;
    #pragma unroll
    for (int dp = 0; dp < 3; ++dp)
        #pragma unroll
        for (int dq = 0; dq < 3; ++dq) {
            int y = 3 * i + dp, x = 3 * j + dq;
            float sv = sin_[y * OSZ + x];
            float iv = ill[y * OSZ + x] * 0.125f;   // /TIME_STEPS
            si += sv * iv;
            ss += sv;
        }
    illum[p] = si;
    sens[p]  = ss * (1.0f / 9.0f);
}

// ---------------- pack weights to f16 [tap][oc][ic] --------------------
__global__ __launch_bounds__(256) void k_prepw(const float* __restrict__ w2,
                                               const float* __restrict__ w3,
                                               _Float16* __restrict__ w2h,
                                               _Float16* __restrict__ w3h) {
    int id = blockIdx.x * blockDim.x + threadIdx.x;
    if (id < 9 * 32 * 32) {
        int tap = id / (32 * 32), r = id % (32 * 32);
        int oc = r / 32, ic = r % 32;
        w2h[id] = (_Float16)w2[(oc * 32 + ic) * 9 + tap];
    }
    if (id < 9 * 16 * 32) {
        int tap = id / (16 * 32), r = id % (16 * 32);
        int oc = r / 32, ic = r % 32;
        w3h[id] = (_Float16)((oc < 12) ? w3[(oc * 32 + ic) * 9 + tap] : 0.0f);
    }
}

// ---------------- charge init + ramp[0] --------------------------------
__global__ __launch_bounds__(256) void k_init(const float* __restrict__ bias,
                                              const float* __restrict__ wsf,
                                              float* __restrict__ charge,
                                              float* __restrict__ ramp0) {
    int p = blockIdx.x * blockDim.x + threadIdx.x;
    if (p >= NS) return;
    float c = bias[p] - wsf[0];
    charge[p] = c;
    ramp0[p]  = c + wsf[1];
}

// ---------------- conv1: 1 -> 32, 3x3 SAME, relu, to f16 [pix][c] ------
__global__ __launch_bounds__(256) void k_conv1(const float* __restrict__ charge,
                                               const float* __restrict__ wsf,
                                               const float* __restrict__ w1,
                                               _Float16* __restrict__ feat1) {
    int p = blockIdx.x * blockDim.x + threadIdx.x;
    if (p >= NS) return;
    int i = p >> 8, j = p & 255;
    float mu = wsf[2];
    float inv[9];
    #pragma unroll
    for (int dy = 0; dy < 3; ++dy)
        #pragma unroll
        for (int dx = 0; dx < 3; ++dx) {
            int y = i + dy - 1, x = j + dx - 1;
            bool ok = (y >= 0) && (y < NP) && (x >= 0) && (x < NP);
            inv[dy * 3 + dx] = ok ? (charge[y * NP + x] - mu) : 0.0f;
        }
    _Float16 out[32];
    #pragma unroll
    for (int oc = 0; oc < 32; ++oc) {
        float s = 0.0f;
        #pragma unroll
        for (int t = 0; t < 9; ++t) s = fmaf(w1[oc * 9 + t], inv[t], s);
        out[oc] = (_Float16)fmaxf(s, 0.0f);
    }
    v8h* dst = (v8h*)(feat1 + (size_t)p * 32);
    #pragma unroll
    for (int k = 0; k < 4; ++k) {
        v8h v;
        #pragma unroll
        for (int e = 0; e < 8; ++e) v[e] = out[k * 8 + e];
        dst[k] = v;
    }
}

// ---------------- conv2: 32 -> 32 implicit-GEMM via WMMA, relu ---------
// One wave handles a 16-pixel row strip x all 32 output channels.
// A (16x32 f16): lane m=l&15 -> pixel x0+m(+dx); K contiguous: half hi holds ic hi*16..+15.
// B (32x16 f16): lane n=l&15 -> oc; halves = 16 contiguous ic.
__global__ __launch_bounds__(256) void k_conv2(const _Float16* __restrict__ feat1,
                                               const _Float16* __restrict__ w2h,
                                               _Float16* __restrict__ feat2) {
    const int wave = blockIdx.x * (blockDim.x >> 5) + (threadIdx.x >> 5);
    const int lane = threadIdx.x & 31;
    const int row  = wave >> 4;          // 16 strips per row of 256 px
    const int x0   = (wave & 15) << 4;
    const int m    = lane & 15;
    const int hi   = lane >> 4;          // lane group 0/1

    v8f acc0 = {}, acc1 = {};
    const v8h zero8 = {};

    #pragma unroll
    for (int dy = 0; dy < 3; ++dy) {
        const int y = row + dy - 1;
        #pragma unroll
        for (int dx = 0; dx < 3; ++dx) {
            const int tap = dy * 3 + dx;
            const int x   = x0 + m + dx - 1;
            const bool ok = (y >= 0) && (y < NP) && (x >= 0) && (x < NP);

            union { v16h v; v8h h[2]; } A;
            if (ok) {
                const _Float16* pp = feat1 + ((size_t)(y * NP + x)) * 32 + hi * 16;
                A.h[0] = *(const v8h*)pp;          // K = hi*16 .. +7
                A.h[1] = *(const v8h*)(pp + 8);    // K = hi*16+8 .. +7
            } else {
                A.h[0] = zero8;
                A.h[1] = zero8;
            }

            const _Float16* wb = w2h + (size_t)tap * 32 * 32 + (lane & 15) * 32 + hi * 16;
            v16h B0 = *(const v16h*)wb;                 // oc tile 0 (0..15)
            v16h B1 = *(const v16h*)(wb + 16 * 32);     // oc tile 1 (16..31)

            acc0 = __builtin_amdgcn_wmma_f32_16x16x32_f16(
                       false, A.v, false, B0, (short)0, acc0, false, false);
            acc1 = __builtin_amdgcn_wmma_f32_16x16x32_f16(
                       false, A.v, false, B1, (short)0, acc1, false, false);
        }
    }

    // C layout: element r -> M = r + hi*8, N = lane&15
    const int n = lane & 15;
    #pragma unroll
    for (int r = 0; r < 8; ++r) {
        const int x = x0 + r + hi * 8;
        _Float16* dp = feat2 + ((size_t)(row * NP + x)) * 32;
        dp[n]      = (_Float16)fmaxf(acc0[r], 0.0f);
        dp[16 + n] = (_Float16)fmaxf(acc1[r], 0.0f);
    }
}

// ---------------- conv3: 32 -> 12 (padded to 16) via WMMA, f32 out -----
__global__ __launch_bounds__(256) void k_conv3(const _Float16* __restrict__ feat2,
                                               const _Float16* __restrict__ w3h,
                                               float* __restrict__ coeff) {
    const int wave = blockIdx.x * (blockDim.x >> 5) + (threadIdx.x >> 5);
    const int lane = threadIdx.x & 31;
    const int row  = wave >> 4;
    const int x0   = (wave & 15) << 4;
    const int m    = lane & 15;
    const int hi   = lane >> 4;

    v8f acc = {};
    const v8h zero8 = {};

    #pragma unroll
    for (int dy = 0; dy < 3; ++dy) {
        const int y = row + dy - 1;
        #pragma unroll
        for (int dx = 0; dx < 3; ++dx) {
            const int tap = dy * 3 + dx;
            const int x   = x0 + m + dx - 1;
            const bool ok = (y >= 0) && (y < NP) && (x >= 0) && (x < NP);

            union { v16h v; v8h h[2]; } A;
            if (ok) {
                const _Float16* pp = feat2 + ((size_t)(y * NP + x)) * 32 + hi * 16;
                A.h[0] = *(const v8h*)pp;
                A.h[1] = *(const v8h*)(pp + 8);
            } else {
                A.h[0] = zero8;
                A.h[1] = zero8;
            }

            const _Float16* wb = w3h + (size_t)tap * 16 * 32 + (lane & 15) * 32 + hi * 16;
            v16h B = *(const v16h*)wb;

            acc = __builtin_amdgcn_wmma_f32_16x16x32_f16(
                      false, A.v, false, B, (short)0, acc, false, false);
        }
    }

    const int n = lane & 15;
    #pragma unroll
    for (int r = 0; r < 8; ++r) {
        const int x = x0 + r + hi * 8;
        coeff[((size_t)(row * NP + x)) * 16 + n] = acc[r];
    }
}

// ---------------- coeffs -> distorted sub-pixel coords (2x3x3/pixel) ---
// POWERS pairs: (0,0)(0,1)(1,0)(0,2)(1,1)(2,0); knots kx=(p-1)/3, ky=(q-1)/3
__global__ __launch_bounds__(256) void k_coords(const float* __restrict__ coeff,
                                                float* __restrict__ coords) {
    int p = blockIdx.x * blockDim.x + threadIdx.x;
    if (p >= NS) return;
    const float* c = coeff + (size_t)p * 16;
    float* o = coords + (size_t)p * 18;
    #pragma unroll
    for (int ch = 0; ch < 2; ++ch) {
        float c0 = c[ch * 6 + 0], c1 = c[ch * 6 + 1], c2 = c[ch * 6 + 2];
        float c3 = c[ch * 6 + 3], c4 = c[ch * 6 + 4], c5 = c[ch * 6 + 5];
        #pragma unroll
        for (int pq = 0; pq < 9; ++pq) {
            int pp = pq / 3, qq = pq % 3;
            float kx = (float)(pp - 1) * (1.0f / 3.0f);
            float ky = (float)(qq - 1) * (1.0f / 3.0f);
            float dist = c0 + c1 * ky + c2 * kx + c3 * ky * ky + c4 * kx * ky + c5 * kx * kx;
            o[ch * 9 + pq] = ((ch == 0) ? kx : ky) + dist;
        }
    }
}

// ---------------- fused calc_kernels + apply + charge update -----------
__global__ __launch_bounds__(256) void k_apply(const float* __restrict__ coords,
                                               const float* __restrict__ illRaw,
                                               const float* __restrict__ illumDS,
                                               const float* __restrict__ sens,
                                               float* __restrict__ charge,
                                               const float* __restrict__ wsf,
                                               float* __restrict__ rampT,
                                               float* __restrict__ diffT) {
    int p = blockIdx.x * blockDim.x + threadIdx.x;
    if (p >= NS) return;
    const int i = p >> 8, j = p & 255;
    float acc = 0.0f;
    #pragma unroll
    for (int u = 0; u < 9; ++u) {
        const int du = u / 3, pp = u % 3;
        const int ii = i - 1 + du;
        const int iy = 3 * (i - 1) + u;          // illuminance row
        #pragma unroll
        for (int v = 0; v < 9; ++v) {
            const int dv = v / 3, qq = v % 3;
            const int jj = j - 1 + dv;
            float wx, wy;
            const bool valid = (ii >= 0) && (ii < NP) && (jj >= 0) && (jj < NP);
            if (valid) {
                const float* cc = coords + ((size_t)(ii * NP + jj)) * 18;
                wx = cc[pp * 3 + qq];
                wy = cc[9 + pp * 3 + qq];
            } else {                              // border: undistorted knots
                wx = (float)(pp - 1) * (1.0f / 3.0f);
                wy = (float)(qq - 1) * (1.0f / 3.0f);
            }
            const float X = wx + (float)(du - 1);
            const float Y = wy + (float)(dv - 1);
            float ox = fminf(X + (1.0f / 6.0f), 0.5f) - fmaxf(X - (1.0f / 6.0f), -0.5f);
            float oy = fminf(Y + (1.0f / 6.0f), 0.5f) - fmaxf(Y - (1.0f / 6.0f), -0.5f);
            ox = fmaxf(ox, 0.0f);
            oy = fmaxf(oy, 0.0f);
            const float frac = ox * oy * 9.0f;    // / (size*size), size=1/3
            const int ix = 3 * (j - 1) + v;
            const float iv = (iy >= 0 && iy < OSZ && ix >= 0 && ix < OSZ)
                                 ? illRaw[iy * OSZ + ix] * 0.125f : 0.0f;
            acc = fmaf(iv, frac, acc);
        }
    }
    const float nc = sens[p] * acc;
    const float c  = charge[p] + nc;
    charge[p] = c;
    rampT[p]  = c + wsf[1];
    diffT[p]  = illumDS[p] - nc;
}

// ---------------- host-side launch orchestration -----------------------
extern "C" void kernel_launch(void* const* d_in, const int* in_sizes, int n_in,
                              void* d_out, int out_size, void* d_ws, size_t ws_size,
                              hipStream_t stream) {
    const float* bias        = (const float*)d_in[0];
    const float* illuminance = (const float*)d_in[1];
    const float* sensitivity = (const float*)d_in[2];
    const float* w1          = (const float*)d_in[3];
    const float* w2          = (const float*)d_in[4];
    const float* w3          = (const float*)d_in[5];

    float* out = (float*)d_out;
    float* wsf = (float*)d_ws;

    // workspace layout (floats then halves; all 32B-aligned chunks)
    float* illum  = wsf + 16;                        // 65536
    float* sens   = illum  + NS;                     // 65536
    float* charge = sens   + NS;                     // 65536
    float* coeff  = charge + NS;                     // 65536*16
    float* coords = coeff  + (size_t)NS * 16;        // 65536*18
    _Float16* feat1 = (_Float16*)(coords + (size_t)NS * 18);  // 65536*32
    _Float16* feat2 = feat1 + (size_t)NS * 32;                // 65536*32
    _Float16* w2h   = feat2 + (size_t)NS * 32;                // 9*32*32
    _Float16* w3h   = w2h + 9 * 32 * 32;                      // 9*16*32

    float* ramp = out;                    // 9 * 65536
    float* diff = out + (size_t)9 * NS;   // 8 * 65536

    const int gPix = (NS + 255) / 256;

    k_stats<<<1, 1024, 0, stream>>>(bias, wsf);
    k_down <<<gPix, 256, 0, stream>>>(illuminance, sensitivity, illum, sens);
    k_prepw<<<(9 * 32 * 32 + 255) / 256, 256, 0, stream>>>(w2, w3, w2h, w3h);
    k_init <<<gPix, 256, 0, stream>>>(bias, wsf, charge, ramp);

    for (int t = 0; t < 8; ++t) {
        k_meanq <<<1, 1024, 0, stream>>>(charge, wsf);
        k_conv1 <<<gPix, 256, 0, stream>>>(charge, wsf, w1, feat1);
        k_conv2 <<<512, 256, 0, stream>>>(feat1, w2h, feat2);
        k_conv3 <<<512, 256, 0, stream>>>(feat2, w3h, coeff);
        k_coords<<<gPix, 256, 0, stream>>>(coeff, coords);
        k_apply <<<gPix, 256, 0, stream>>>(coords, illuminance, illum, sens, charge,
                                           wsf, ramp + (size_t)(t + 1) * NS,
                                           diff + (size_t)t * NS);
    }
}
